// Baseline_new_15229954031625
// MI455X (gfx1250) — matrix operand order repaired and checked by reading, not compiled, not run on hardware
//
#include <hip/hip_runtime.h>
#include <hip/hip_bf16.h>

typedef __attribute__((ext_vector_type(16))) _Float16 v16h;
typedef __attribute__((ext_vector_type(8)))  _Float16 v8h;
typedef __attribute__((ext_vector_type(8)))  float    v8f;

#define B_   2048
#define L_   512
#define D_   15
#define NM_  32
#define F_   40
#define FH_  20
#define FC_K 10240   // FH_ * L_
#define WSTR 40      // packed-weight / LDS fragment row stride (f16 elems)

__device__ __forceinline__ v16h cat16(v8h lo, v8h hi) {
    return __builtin_shufflevector(lo, hi, 0, 1, 2, 3, 4, 5, 6, 7,
                                   8, 9, 10, 11, 12, 13, 14, 15);
}

// =====================================================================
// Generic 1-D conv as implicit-im2col WMMA GEMM.
//   out[b,co,l] = act( sum_{ci,kk} in[b,ci,l+kk-pad]*W + bias[co] + skip )
// Weights pre-packed f16 in FRAGMENT order: Wpk[(kc*48 + co)*WSTR + klocal],
// klocal in [0,32), taps padded to 8; all pad entries are zero so the
// MAC loop needs no masking. Tile: 64 positions x (NT*16) channels.
// skip_mode: 0=none, 1=+skip[b,co,l], 2=+wsk[co]*x1[b,l]+bsk[co]
// =====================================================================
template <int NT>
__global__ __launch_bounds__(128) void conv_gemm_kernel(
    const float* __restrict__ in, const float* __restrict__ skip,
    const float* __restrict__ x1, const float* __restrict__ wsk,
    const float* __restrict__ bsk,
    const _Float16* __restrict__ Wpk, const float* __restrict__ bias,
    float* __restrict__ out,
    int Cin, int Cout, int taps, int pad, int relu, int skip_mode)
{
    __shared__ __align__(16) _Float16 sIn[F_ * 72];          // input + halo
    __shared__ __align__(16) _Float16 sW[10 * 48 * WSTR];    // 38400 B
    __shared__ float sX[64];
    __shared__ float sBias[48], sWsk[48], sBsk[48];

    const int tid = threadIdx.x;
    const int b   = blockIdx.x >> 3;        // L_/64 = 8 tiles per batch row
    const int l0  = (blockIdx.x & 7) * 64;

    const int kChunks = (Cin * 8 + 31) >> 5;
    const int CinPad  = kChunks * 4;        // rows touched by A fragments

    // weights: straight vector copy (already zero-padded in global)
    {
        const unsigned* src = (const unsigned*)Wpk;
        unsigned*       dst = (unsigned*)sW;
        const int nw = kChunks * 48 * (WSTR / 2);
        for (int idx = tid; idx < nw; idx += 128) dst[idx] = src[idx];
    }
    // input tile: fill ALL CinPad*72 entries (zero rows >= Cin, zero OOB l)
    for (int idx = tid; idx < CinPad * 72; idx += 128) {
        int ci = idx / 72, p = idx - ci * 72;
        int l  = l0 - pad + p;
        float v = 0.0f;
        if (ci < Cin && l >= 0 && l < L_) v = in[((size_t)b * Cin + ci) * L_ + l];
        sIn[idx] = (_Float16)v;
    }
    if (tid < 48) {
        sBias[tid] = (tid < Cout) ? bias[tid] : 0.0f;
        if (skip_mode == 2) {
            sWsk[tid] = (tid < Cout) ? wsk[tid] : 0.0f;
            sBsk[tid] = (tid < Cout) ? bsk[tid] : 0.0f;
        }
    }
    if (skip_mode == 2 && tid < 64) sX[tid] = x1[(size_t)b * L_ + l0 + tid];
    __syncthreads();

    const int wave = tid >> 5, lane = tid & 31;
    const int lh = lane >> 4, lr = lane & 15;
    const int pos = wave * 16 + lr;         // this lane's A row (position)

    v8f acc[NT];
    #pragma unroll
    for (int nt = 0; nt < NT; nt++)
        #pragma unroll
        for (int i = 0; i < 8; i++) acc[nt][i] = 0.0f;

    for (int kc = 0; kc < kChunks; kc++) {
        // A fragment: two contiguous 8xf16 runs (rows ci0 and ci0+2)
        const int ci0 = kc * 4 + lh;
        v16h a;
        #pragma unroll
        for (int i = 0; i < 8; i++) {
            a[i]     = sIn[ci0 * 72 + pos + i];
            a[i + 8] = sIn[(ci0 + 2) * 72 + pos + i];
        }
        #pragma unroll
        for (int nt = 0; nt < NT; nt++) {
            const _Float16* wp = &sW[(kc * 48 + nt * 16 + lr) * WSTR + lh * 8];
            v16h bf = cat16(*(const v8h*)wp, *(const v8h*)(wp + 16));
            acc[nt] = __builtin_amdgcn_wmma_f32_16x16x32_f16(
                false, a, false, bf, (short)0, acc[nt], false, false);
        }
    }

    // epilogue: bias + skip + relu, store fp32 (B, Cout, L)
    #pragma unroll
    for (int nt = 0; nt < NT; nt++) {
        int co = nt * 16 + lr;
        if (co >= Cout) continue;
        #pragma unroll
        for (int r = 0; r < 8; r++) {
            int m = wave * 16 + r + 8 * lh;   // C/D layout: M = r + 8*(lane/16)
            int l = l0 + m;
            float v = acc[nt][r] + sBias[co];
            if (skip_mode == 1)      v += skip[((size_t)b * Cout + co) * L_ + l];
            else if (skip_mode == 2) v += sWsk[co] * sX[m] + sBsk[co];
            if (relu) v = fmaxf(v, 0.0f);
            out[((size_t)b * Cout + co) * L_ + l] = v;
        }
    }
}

// =====================================================================
// VQ: distances to 32 means, softmax -> prob (B,L,32), q = prob @ means^T
// =====================================================================
__global__ __launch_bounds__(256) void vq_kernel(
    const float* __restrict__ code, const float* __restrict__ means,
    float* __restrict__ prob, float* __restrict__ q)
{
    __shared__ float sM[D_ * NM_];
    __shared__ float sM2[NM_];
    int tid = threadIdx.x;
    for (int i = tid; i < D_ * NM_; i += 256) sM[i] = means[i];
    __syncthreads();
    if (tid < NM_) {
        float s = 0.0f;
        for (int d = 0; d < D_; d++) { float v = sM[d * NM_ + tid]; s += v * v; }
        sM2[tid] = s;
    }
    __syncthreads();

    size_t idx = (size_t)blockIdx.x * 256 + tid;
    if (idx >= (size_t)B_ * L_) return;
    int b = (int)(idx >> 9), l = (int)(idx & (L_ - 1));

    float c[D_], c2 = 0.0f;
    #pragma unroll
    for (int d = 0; d < D_; d++) {
        c[d] = code[((size_t)b * D_ + d) * L_ + l];
        c2 += c[d] * c[d];
    }
    float p[NM_], smax = -1e30f;
    #pragma unroll
    for (int m = 0; m < NM_; m++) {
        float dot = 0.0f;
        #pragma unroll
        for (int d = 0; d < D_; d++) dot += c[d] * sM[d * NM_ + m];
        float s = -(c2 + sM2[m] - 2.0f * dot);   // SCALE = 1
        p[m] = s; smax = fmaxf(smax, s);
    }
    float sum = 0.0f;
    #pragma unroll
    for (int m = 0; m < NM_; m++) { p[m] = __expf(p[m] - smax); sum += p[m]; }
    float inv = 1.0f / sum;
    #pragma unroll
    for (int m = 0; m < NM_; m++) {
        p[m] *= inv;
        prob[idx * NM_ + m] = p[m];
    }
    #pragma unroll
    for (int d = 0; d < D_; d++) {
        float qd = 0.0f;
        #pragma unroll
        for (int m = 0; m < NM_; m++) qd += p[m] * sM[d * NM_ + m];
        q[((size_t)b * D_ + d) * L_ + l] = qd;
    }
}

// =====================================================================
// FC: out(2048x512) = tanh(A(2048x10240) @ Bt(512x10240)^T + bias)
// A is [m][k] f16, Bt is [n][k] f16 (fc_w native layout, f16-cast).
// 128x64 tile / 256 threads / 8 waves x (2x2) WMMA tiles. All fragments
// load as pairs of 16B-aligned LDS reads (row stride 40 f16 = 80B).
// =====================================================================
__global__ __launch_bounds__(256) void fc_kernel(
    const _Float16* __restrict__ A, const _Float16* __restrict__ Bt,
    const float* __restrict__ bias, float* __restrict__ out)
{
    __shared__ __align__(16) _Float16 As[128 * WSTR];   // 10240 B
    __shared__ __align__(16) _Float16 Bs[64 * WSTR];    //  5120 B

    const int tid = threadIdx.x;
    const int m0 = blockIdx.x * 128;
    const int n0 = blockIdx.y * 64;
    const int wave = tid >> 5, lane = tid & 31;
    const int lh = lane >> 4, lr = lane & 15;
    const int wm = wave & 3, wn = wave >> 2;

    v8f acc[2][2];
    #pragma unroll
    for (int mt = 0; mt < 2; mt++)
        #pragma unroll
        for (int nt = 0; nt < 2; nt++)
            #pragma unroll
            for (int i = 0; i < 8; i++) acc[mt][nt][i] = 0.0f;

    for (int kc = 0; kc < FC_K / 32; kc++) {
        // A tile 128x32 and B tile 64x32, u32-vectorized copies
        for (int idx = tid; idx < 128 * 16; idx += 256) {
            int r = idx >> 4, c2 = idx & 15;
            const _Float16* src = &A[((size_t)(m0 + r)) * FC_K + kc * 32 + c2 * 2];
            __builtin_prefetch(src + 32, 0, 1);
            *(unsigned*)&As[r * WSTR + c2 * 2] = *(const unsigned*)src;
        }
        for (int idx = tid; idx < 64 * 16; idx += 256) {
            int r = idx >> 4, c2 = idx & 15;
            *(unsigned*)&Bs[r * WSTR + c2 * 2] =
                *(const unsigned*)&Bt[((size_t)(n0 + r)) * FC_K + kc * 32 + c2 * 2];
        }
        __syncthreads();

        v16h af[2], bf[2];
        #pragma unroll
        for (int mt = 0; mt < 2; mt++) {
            const _Float16* p = &As[(wm * 32 + mt * 16 + lr) * WSTR + 8 * lh];
            af[mt] = cat16(*(const v8h*)p, *(const v8h*)(p + 16));
        }
        #pragma unroll
        for (int nt = 0; nt < 2; nt++) {
            const _Float16* p = &Bs[(wn * 32 + nt * 16 + lr) * WSTR + 8 * lh];
            bf[nt] = cat16(*(const v8h*)p, *(const v8h*)(p + 16));
        }
        #pragma unroll
        for (int mt = 0; mt < 2; mt++)
            #pragma unroll
            for (int nt = 0; nt < 2; nt++)
                acc[mt][nt] = __builtin_amdgcn_wmma_f32_16x16x32_f16(
                    false, af[mt], false, bf[nt], (short)0, acc[mt][nt], false, false);
        __syncthreads();
    }

    #pragma unroll
    for (int mt = 0; mt < 2; mt++)
        #pragma unroll
        for (int nt = 0; nt < 2; nt++) {
            int n = n0 + wn * 32 + nt * 16 + lr;
            float bn = bias[n];
            #pragma unroll
            for (int r = 0; r < 8; r++) {
                int m = m0 + wm * 32 + mt * 16 + r + 8 * lh;
                out[(size_t)m * L_ + n] = tanhf(acc[mt][nt][r] + bn);
            }
        }
}

// =====================================================================
// Weight packing / casting helpers
// =====================================================================
// Pack conv weights to fragment-order f16: dst[(kc*48+co)*WSTR + klocal].
// Zeros for klocal>=32, co>=Co, ci>=Ci, kk>=Kt (taps padded 8).
__global__ void pack_conv_kernel(const float* __restrict__ w, _Float16* __restrict__ dst,
                                 int Co, int Ci, int Kt)
{
    int kChunks = (Ci * 8 + 31) >> 5;
    int n = kChunks * 48 * WSTR;
    for (int idx = blockIdx.x * 256 + threadIdx.x; idx < n; idx += gridDim.x * 256) {
        int row = idx / WSTR, kl = idx - row * WSTR;
        int kc = row / 48, co = row - kc * 48;
        float v = 0.0f;
        if (kl < 32 && co < Co) {
            int k = kc * 32 + kl;
            int ci = k >> 3, kk = k & 7;
            if (ci < Ci && kk < Kt) v = w[((size_t)co * Ci + ci) * Kt + kk];
        }
        dst[idx] = (_Float16)v;
    }
}

__global__ void cast_f16_kernel(const float* __restrict__ src, _Float16* __restrict__ dst, size_t n)
{
    for (size_t idx = (size_t)blockIdx.x * 256 + threadIdx.x; idx < n;
         idx += (size_t)gridDim.x * 256)
        dst[idx] = (_Float16)src[idx];
}

// =====================================================================
extern "C" void kernel_launch(void* const* d_in, const int* in_sizes, int n_in,
                              void* d_out, int out_size, void* d_ws, size_t ws_size,
                              hipStream_t stream)
{
    (void)in_sizes; (void)n_in; (void)out_size; (void)ws_size;
    const float* x      = (const float*)d_in[0];
    const float* means  = (const float*)d_in[1];
    const float* eb_w1  = (const float*)d_in[2];
    const float* eb_b1  = (const float*)d_in[3];
    const float* eb_w2  = (const float*)d_in[4];
    const float* eb_b2  = (const float*)d_in[5];
    const float* eb_wsk = (const float*)d_in[6];
    const float* eb_bsk = (const float*)d_in[7];
    const float* ec_w   = (const float*)d_in[8];
    const float* ec_b   = (const float*)d_in[9];
    const float* di_w   = (const float*)d_in[10];
    const float* di_b   = (const float*)d_in[11];
    const float* fc_w   = (const float*)d_in[12];
    const float* fc_b   = (const float*)d_in[13];
    const float* bn1_w1 = (const float*)d_in[14];
    const float* bn1_b1 = (const float*)d_in[15];
    const float* bn1_w2 = (const float*)d_in[16];
    const float* bn1_b2 = (const float*)d_in[17];
    const float* bn1_w3 = (const float*)d_in[18];
    const float* bn1_b3 = (const float*)d_in[19];
    const float* bn2_w1 = (const float*)d_in[20];
    const float* bn2_b1 = (const float*)d_in[21];
    const float* bn2_w2 = (const float*)d_in[22];
    const float* bn2_b2 = (const float*)d_in[23];
    const float* bn2_w3 = (const float*)d_in[24];
    const float* bn2_b3 = (const float*)d_in[25];
    const float* dib_w1 = (const float*)d_in[26];
    const float* dib_b1 = (const float*)d_in[27];
    const float* dib_w2 = (const float*)d_in[28];
    const float* dib_b2 = (const float*)d_in[29];
    const float* dib_w3 = (const float*)d_in[30];
    const float* dib_b3 = (const float*)d_in[31];
    const float* do1_w1 = (const float*)d_in[32];
    const float* do1_b1 = (const float*)d_in[33];
    const float* do1_w2 = (const float*)d_in[34];
    const float* do1_b2 = (const float*)d_in[35];
    const float* do2_w1 = (const float*)d_in[36];
    const float* do2_b1 = (const float*)d_in[37];
    const float* do2_w2 = (const float*)d_in[38];
    const float* do2_b2 = (const float*)d_in[39];

    // ---- workspace bump allocator ----
    char* ws = (char*)d_ws;
    size_t off = 0;
    auto alloc = [&](size_t bytes) -> void* {
        void* p = ws + off;
        off = (off + bytes + 255) & ~(size_t)255;
        return p;
    };
    auto a16 = [&](int Ci) {
        int kChunks = (Ci * 8 + 31) >> 5;
        return (_Float16*)alloc((size_t)kChunks * 48 * WSTR * 2);
    };

    _Float16* wp_eb1  = a16(1);
    _Float16* wp_eb2  = a16(F_);
    _Float16* wp_ec   = a16(F_);
    _Float16* wp_di   = a16(D_);
    _Float16* wp_b11  = a16(D_); _Float16* wp_b12 = a16(D_); _Float16* wp_b13 = a16(D_);
    _Float16* wp_b21  = a16(D_); _Float16* wp_b22 = a16(D_); _Float16* wp_b23 = a16(D_);
    _Float16* wp_db1  = a16(FH_); _Float16* wp_db2 = a16(FH_); _Float16* wp_db3 = a16(FH_);
    _Float16* wp_d11  = a16(FH_); _Float16* wp_d12 = a16(FH_);
    _Float16* wp_d21  = a16(FH_); _Float16* wp_d22 = a16(FH_);
    _Float16* fcw16   = (_Float16*)alloc((size_t)L_ * FC_K * 2);
    _Float16* A16     = (_Float16*)alloc((size_t)B_ * FC_K * 2);
    float* big0 = (float*)alloc((size_t)B_ * F_ * L_ * 4);
    float* big1 = (float*)alloc((size_t)B_ * F_ * L_ * 4);
    float* s0 = big0;
    float* s1 = big0 + (size_t)B_ * FH_ * L_;
    float* s2 = big1;
    float* s3 = big1 + (size_t)B_ * FH_ * L_;

    // ---- pack weights to f16 fragment layout ----
    auto packc = [&](const float* w, _Float16* dst, int Co, int Ci, int Kt) {
        int kChunks = (Ci * 8 + 31) >> 5;
        int n = kChunks * 48 * WSTR;
        pack_conv_kernel<<<(n + 255) / 256, 256, 0, stream>>>(w, dst, Co, Ci, Kt);
    };
    packc(eb_w1, wp_eb1, F_, 1, 5);
    packc(eb_w2, wp_eb2, F_, F_, 5);
    packc(ec_w,  wp_ec,  D_, F_, 5);
    packc(di_w,  wp_di,  FH_, D_, 3);
    packc(bn1_w1, wp_b11, D_, D_, 1); packc(bn1_w2, wp_b12, D_, D_, 5); packc(bn1_w3, wp_b13, D_, D_, 1);
    packc(bn2_w1, wp_b21, D_, D_, 1); packc(bn2_w2, wp_b22, D_, D_, 5); packc(bn2_w3, wp_b23, D_, D_, 1);
    packc(dib_w1, wp_db1, FH_, FH_, 1); packc(dib_w2, wp_db2, FH_, FH_, 5); packc(dib_w3, wp_db3, FH_, FH_, 1);
    packc(do1_w1, wp_d11, FH_, FH_, 5); packc(do1_w2, wp_d12, FH_, FH_, 5);
    packc(do2_w1, wp_d21, FH_, FH_, 5); packc(do2_w2, wp_d22, FH_, FH_, 5);
    // fc_w is already [n][k]; just cast to f16
    cast_f16_kernel<<<8192, 256, 0, stream>>>(fc_w, fcw16, (size_t)L_ * FC_K);

    // ---- conv launcher (NT specialization by Cout) ----
    const dim3 cgrid(B_ * (L_ / 64));
    auto conv = [&](const float* in, const float* skip, const float* wsk, const float* bsk,
                    const _Float16* Wpk, const float* bias, float* out,
                    int Cin, int Cout, int taps, int pad, int relu, int skip_mode) {
        if (Cout <= 16)
            conv_gemm_kernel<1><<<cgrid, 128, 0, stream>>>(in, skip, x, wsk, bsk, Wpk, bias,
                                                           out, Cin, Cout, taps, pad, relu, skip_mode);
        else if (Cout <= 32)
            conv_gemm_kernel<2><<<cgrid, 128, 0, stream>>>(in, skip, x, wsk, bsk, Wpk, bias,
                                                           out, Cin, Cout, taps, pad, relu, skip_mode);
        else
            conv_gemm_kernel<3><<<cgrid, 128, 0, stream>>>(in, skip, x, wsk, bsk, Wpk, bias,
                                                           out, Cin, Cout, taps, pad, relu, skip_mode);
    };

    // encoder basic block (eb): t0 = relu(conv1(x)); h1 = relu(conv2(t0) + wsk*x+bsk)
    conv(x,    nullptr, nullptr, nullptr, wp_eb1, eb_b1, big0, 1,  F_, 5, 2, 1, 0);
    conv(big0, nullptr, eb_wsk,  eb_bsk,  wp_eb2, eb_b2, big1, F_, F_, 5, 2, 1, 2);
    // encoder conv (ec): 40 -> 15, no relu
    conv(big1, nullptr, nullptr, nullptr, wp_ec, ec_b, s0, F_, D_, 5, 2, 0, 0);
    // bottleneck bn1
    conv(s0, nullptr, nullptr, nullptr, wp_b11, bn1_b1, s2, D_, D_, 1, 0, 1, 0);
    conv(s2, nullptr, nullptr, nullptr, wp_b12, bn1_b2, s3, D_, D_, 5, 2, 1, 0);
    conv(s3, s0,      nullptr, nullptr, wp_b13, bn1_b3, s1, D_, D_, 1, 0, 1, 1);
    // bottleneck bn2
    conv(s1, nullptr, nullptr, nullptr, wp_b21, bn2_b1, s2, D_, D_, 1, 0, 1, 0);
    conv(s2, nullptr, nullptr, nullptr, wp_b22, bn2_b2, s3, D_, D_, 5, 2, 1, 0);
    conv(s3, s1,      nullptr, nullptr, wp_b23, bn2_b3, s0, D_, D_, 1, 0, 1, 1);
    // VQ softmax: prob -> second output region; q -> s1
    float* prob_out = (float*)d_out + (size_t)B_ * L_;
    vq_kernel<<<(B_ * L_ + 255) / 256, 256, 0, stream>>>(s0, means, prob_out, s1);
    // decoder input conv (di): 15 -> 20, k3, relu
    conv(s1, nullptr, nullptr, nullptr, wp_di, di_b, s2, D_, FH_, 3, 1, 1, 0);
    // bottleneck dib
    conv(s2, nullptr, nullptr, nullptr, wp_db1, dib_b1, s3, FH_, FH_, 1, 0, 1, 0);
    conv(s3, nullptr, nullptr, nullptr, wp_db2, dib_b2, s0, FH_, FH_, 5, 2, 1, 0);
    conv(s0, s2,      nullptr, nullptr, wp_db3, dib_b3, s1, FH_, FH_, 1, 0, 1, 1);
    // basic block do1 (identity skip)
    conv(s1, nullptr, nullptr, nullptr, wp_d11, do1_b1, s2, FH_, FH_, 5, 2, 1, 0);
    conv(s2, s1,      nullptr, nullptr, wp_d12, do1_b2, s3, FH_, FH_, 5, 2, 1, 1);
    // basic block do2
    conv(s3, nullptr, nullptr, nullptr, wp_d21, do2_b1, s0, FH_, FH_, 5, 2, 1, 0);
    conv(s0, s3,      nullptr, nullptr, wp_d22, do2_b2, s1, FH_, FH_, 5, 2, 1, 1);
    // flatten (B,20,L) -> f16 [B][10240] (same linear layout) and FC + tanh
    cast_f16_kernel<<<8192, 256, 0, stream>>>(s1, A16, (size_t)B_ * FH_ * L_);
    fc_kernel<<<dim3(B_ / 128, L_ / 64), 256, 0, stream>>>(A16, fcw16, fc_b, (float*)d_out);
}
